// MPNNTransform_90898687853303
// MI455X (gfx1250) — matrix operand order repaired and checked by reading, not compiled, not run on hardware
//
#include <hip/hip_runtime.h>
#include <hip/hip_bf16.h>

// MI455X / gfx1250: wave32, WMMA 16x16x32 bf16 -> f32 accum.
// ~180 GFLOP of GEMMs on an L2-resident (~45MB) working set -> compute bound.
// v2: 2x4 tile block per wave (better A/B reuse) + double-buffered k-pipeline
// so global_load_b128 of the next k-step overlaps the current 8 WMMAs
// (avoids the s_wait_loadcnt 0 before every WMMA seen in v1).

typedef __attribute__((ext_vector_type(16))) __bf16 v16bf;
typedef __attribute__((ext_vector_type(8)))  __bf16 v8bf;
typedef __attribute__((ext_vector_type(8)))  float  v8f;

#define DEV __device__ __forceinline__

constexpr int BATCH = 64;
constexpr int NNODE = 256;
constexpr int HDIM  = 256;
constexpr int FPAD  = 32;    // F=16 padded to one K-step of 32
constexpr int NITER = 10;

union ABfrag { v16bf v; v8bf h[2]; };

// A fragment: 16x32 (MxK), row-major source, ld == K (row length).
// lanes 0-15 -> M=lane, elems 0..7 = K 0..7, elems 8..15 = K 16..23
// lanes 16-31 -> elems 0..7 = K 8..15, elems 8..15 = K 24..31
DEV v16bf load_a_frag(const __bf16* __restrict__ base, int row, int ld, int k, int lane) {
    int m  = lane & 15;
    int kh = lane >> 4;
    const __bf16* p = base + (size_t)(row + m) * (size_t)ld + k + kh * 8;
    ABfrag f;
    f.h[0] = *(const v8bf*)(p);
    f.h[1] = *(const v8bf*)(p + 16);
    return f.v;
}

// B fragment: 32x16 (KxN), source stored TRANSPOSED (B^T row-major, [N][K]).
// lane n = lane&15; lanes 0-15: elems 0..7 = K 0..7, 8..15 = K 16..23; lanes 16-31: K 8..15, 24..31.
DEV v16bf load_b_frag(const __bf16* __restrict__ baseT, int col, int ld, int k, int lane) {
    int n  = lane & 15;
    int kh = lane >> 4;
    const __bf16* p = baseT + (size_t)(col + n) * (size_t)ld + k + kh * 8;
    ABfrag f;
    f.h[0] = *(const v8bf*)(p);
    f.h[1] = *(const v8bf*)(p + 16);
    return f.v;
}

DEV float sigmoidf_(float x) { return 1.0f / (1.0f + expf(-x)); }

enum { EPI_TSTORE = 0, EPI_ROW_TANH = 1, EPI_ROW_SIG = 2, EPI_ROW_SIGMULH = 3, EPI_ROW_GRU = 4 };

// One k-step's worth of operand fragments for a 2x4 tile block.
struct KFrags { v16bf a0, a1, b0, b1, b2, b3; };

DEV KFrags load_kfrags(const __bf16* __restrict__ A, const __bf16* __restrict__ B,
                       int trow0, int tcol0, int K, int k, int lane) {
    KFrags f;
    f.a0 = load_a_frag(A, trow0,      K, k, lane);
    f.a1 = load_a_frag(A, trow0 + 16, K, k, lane);
    f.b0 = load_b_frag(B, tcol0,      K, k, lane);
    f.b1 = load_b_frag(B, tcol0 + 16, K, k, lane);
    f.b2 = load_b_frag(B, tcol0 + 32, K, k, lane);
    f.b3 = load_b_frag(B, tcol0 + 48, K, k, lane);
    return f;
}

DEV void apply_kfrags(const KFrags& f, v8f acc[2][4]) {
    acc[0][0] = __builtin_amdgcn_wmma_f32_16x16x32_bf16(false, f.a0, false, f.b0, (short)0, acc[0][0], false, false);
    acc[0][1] = __builtin_amdgcn_wmma_f32_16x16x32_bf16(false, f.a0, false, f.b1, (short)0, acc[0][1], false, false);
    acc[0][2] = __builtin_amdgcn_wmma_f32_16x16x32_bf16(false, f.a0, false, f.b2, (short)0, acc[0][2], false, false);
    acc[0][3] = __builtin_amdgcn_wmma_f32_16x16x32_bf16(false, f.a0, false, f.b3, (short)0, acc[0][3], false, false);
    acc[1][0] = __builtin_amdgcn_wmma_f32_16x16x32_bf16(false, f.a1, false, f.b0, (short)0, acc[1][0], false, false);
    acc[1][1] = __builtin_amdgcn_wmma_f32_16x16x32_bf16(false, f.a1, false, f.b1, (short)0, acc[1][1], false, false);
    acc[1][2] = __builtin_amdgcn_wmma_f32_16x16x32_bf16(false, f.a1, false, f.b2, (short)0, acc[1][2], false, false);
    acc[1][3] = __builtin_amdgcn_wmma_f32_16x16x32_bf16(false, f.a1, false, f.b3, (short)0, acc[1][3], false, false);
}

// Generic fused GEMM: Out = epi( sum_p A_p @ B_p + bias ), all M=N_out=256.
// Grid: BATCH*4 workgroups (4 blocks of 128x128 per batch), 256 threads (8 waves).
// Wave tiling: 4x2 wave grid; each wave owns a 32x64 sub-block (2x4 tiles).
template <int EPI, int NSRC>
__global__ __launch_bounds__(256) void gemm_mpnn(
    const __bf16* __restrict__ A0, size_t a0s, const __bf16* __restrict__ B0, size_t b0s, int k0,
    const __bf16* __restrict__ A1, size_t a1s, const __bf16* __restrict__ B1, size_t b1s, int k1,
    const __bf16* __restrict__ A2, size_t a2s, const __bf16* __restrict__ B2, size_t b2s, int k2,
    const float* __restrict__ bias,                     // per-output-column, may be null
    const __bf16* __restrict__ Hbuf,                    // h for SIGMULH / GRU epilogues
    const __bf16* __restrict__ Zbuf,                    // z for GRU epilogue
    size_t hzs,                                         // batch stride for Hbuf/Zbuf
    __bf16* __restrict__ Out, size_t outs)
{
    const int wg    = blockIdx.x;
    const int b     = wg >> 2;
    const int blk   = wg & 3;
    const int brow  = (blk >> 1) * 128;
    const int bcol  = (blk & 1) * 128;
    const int wave  = threadIdx.x >> 5;
    const int lane  = threadIdx.x & 31;
    const int trow0 = brow + (wave & 3) * 32;   // 2 tile rows: trow0, trow0+16
    const int tcol0 = bcol + (wave >> 2) * 64;  // 4 tile cols: tcol0 + 16*c

    v8f acc[2][4] = {};

    auto do_pair = [&](const __bf16* __restrict__ A, size_t as,
                       const __bf16* __restrict__ B, size_t bs, int K) {
        const __bf16* Ab = A + (size_t)b * as;
        const __bf16* Bb = B + (size_t)b * bs;
        KFrags cur = load_kfrags(Ab, Bb, trow0, tcol0, K, 0, lane);
        for (int k = 32; k < K; k += 32) {
            KFrags nxt = load_kfrags(Ab, Bb, trow0, tcol0, K, k, lane);  // overlaps WMMAs below
            apply_kfrags(cur, acc);
            cur = nxt;
        }
        apply_kfrags(cur, acc);
    };

    do_pair(A0, a0s, B0, b0s, k0);
    if constexpr (NSRC > 1) do_pair(A1, a1s, B1, b1s, k1);
    if constexpr (NSRC > 2) do_pair(A2, a2s, B2, b2s, k2);

    // C/D layout: lane n = lane&15; VGPR j -> M = (lane>>4)*8 + j
    const int n  = lane & 15;
    const int kh = lane >> 4;
    #pragma unroll
    for (int r = 0; r < 2; ++r) {
        #pragma unroll
        for (int c = 0; c < 4; ++c) {
            const int ocol = tcol0 + c * 16 + n;
            const int orow = trow0 + r * 16 + kh * 8;
            const float bi = bias ? bias[ocol] : 0.0f;
            if constexpr (EPI == EPI_TSTORE) {
                // store transposed (Out is [HDIM][NNODE] per batch) -> contiguous 16B
                v8bf pk;
                #pragma unroll
                for (int j = 0; j < 8; ++j) pk[j] = (__bf16)(acc[r][c][j] + bi);
                *(v8bf*)(Out + (size_t)b * outs + (size_t)ocol * NNODE + orow) = pk;
            } else if constexpr (EPI == EPI_ROW_TANH) {
                #pragma unroll
                for (int j = 0; j < 8; ++j)
                    Out[(size_t)b * outs + (size_t)(orow + j) * HDIM + ocol] =
                        (__bf16)tanhf(acc[r][c][j] + bi);
            } else if constexpr (EPI == EPI_ROW_SIG) {
                #pragma unroll
                for (int j = 0; j < 8; ++j)
                    Out[(size_t)b * outs + (size_t)(orow + j) * HDIM + ocol] =
                        (__bf16)sigmoidf_(acc[r][c][j] + bi);
            } else if constexpr (EPI == EPI_ROW_SIGMULH) {
                #pragma unroll
                for (int j = 0; j < 8; ++j) {
                    size_t idx = (size_t)b * hzs + (size_t)(orow + j) * HDIM + ocol;
                    float hv = (float)Hbuf[idx];
                    Out[(size_t)b * outs + (size_t)(orow + j) * HDIM + ocol] =
                        (__bf16)(sigmoidf_(acc[r][c][j] + bi) * hv);
                }
            } else {  // EPI_ROW_GRU: h = (1-z)*h + z*tanh(acc+bh)
                #pragma unroll
                for (int j = 0; j < 8; ++j) {
                    size_t idx = (size_t)b * hzs + (size_t)(orow + j) * HDIM + ocol;
                    float ht = tanhf(acc[r][c][j] + bi);
                    float zv = (float)Zbuf[idx];
                    float hv = (float)Hbuf[idx];
                    Out[(size_t)b * outs + (size_t)(orow + j) * HDIM + ocol] =
                        (__bf16)((1.0f - zv) * hv + zv * ht);
                }
            }
        }
    }
}

// ---------------- prep / tail kernels ----------------

__global__ void k_transposeW(const float* __restrict__ src, __bf16* __restrict__ dst) {
    int r = blockIdx.x, c = threadIdx.x;  // 256 x 256
    dst[(size_t)c * 256 + r] = (__bf16)src[(size_t)r * 256 + c];
}

// rows 256..271 of a (272,256) matrix -> B^T [256][32] padded with zeros
__global__ void k_botT(const float* __restrict__ src, __bf16* __restrict__ dst) {
    int i = blockIdx.x * blockDim.x + threadIdx.x;  // 256*32
    int c = i >> 5, f = i & 31;
    dst[i] = (f < 16) ? (__bf16)src[(size_t)(256 + f) * 256 + c] : (__bf16)0.0f;
}

__global__ void k_jets_pad(const float* __restrict__ jets, __bf16* __restrict__ dst) {
    int i = blockIdx.x * blockDim.x + threadIdx.x;  // 64*256*32
    int bn = i >> 5, f = i & 31;
    dst[i] = (f < 16) ? (__bf16)jets[(size_t)bn * 16 + f] : (__bf16)0.0f;
}

__global__ void k_embed(const float* __restrict__ jets, const float* __restrict__ W_emb,
                        const float* __restrict__ b_emb, __bf16* __restrict__ h) {
    int bn = blockIdx.x;   // b*256+n
    int c  = threadIdx.x;  // H
    float s = b_emb[c];
    #pragma unroll
    for (int f = 0; f < 16; ++f) s += jets[(size_t)bn * 16 + f] * W_emb[(size_t)f * 256 + c];
    h[(size_t)bn * 256 + c] = (__bf16)tanhf(s);
}

__global__ void k_tobf(const float* __restrict__ src, __bf16* __restrict__ dst, int ntot) {
    int i = blockIdx.x * blockDim.x + threadIdx.x;
    if (i < ntot) dst[i] = (__bf16)src[i];
}

__global__ void k_reduce(const __bf16* __restrict__ g, float* __restrict__ s) {
    int b = blockIdx.x, k = threadIdx.x;
    float acc = 0.0f;
    for (int n = 0; n < 256; ++n) acc += (float)g[((size_t)b * 256 + n) * 256 + k];
    s[b * 256 + k] = acc;
}

__global__ void k_final(const float* __restrict__ s, const float* __restrict__ W_r2,
                        const float* __restrict__ b_r2, float* __restrict__ out) {
    int b = blockIdx.x, j = threadIdx.x;
    float acc = 256.0f * b_r2[j];
    for (int k = 0; k < 256; ++k) acc += s[b * 256 + k] * W_r2[(size_t)k * 256 + j];
    out[b * 256 + j] = acc;
}

// ---------------- host launcher ----------------

extern "C" void kernel_launch(void* const* d_in, const int* in_sizes, int n_in,
                              void* d_out, int out_size, void* d_ws, size_t ws_size,
                              hipStream_t stream) {
    const float* jets  = (const float*)d_in[0];
    const float* dads  = (const float*)d_in[1];
    const float* W_emb = (const float*)d_in[2];
    const float* b_emb = (const float*)d_in[3];
    const float* W_msg = (const float*)d_in[4];
    const float* b_msg = (const float*)d_in[5];
    const float* Wz    = (const float*)d_in[6];
    const float* Uz    = (const float*)d_in[7];
    const float* bz    = (const float*)d_in[8];
    const float* Wr    = (const float*)d_in[9];
    const float* Ur    = (const float*)d_in[10];
    const float* br    = (const float*)d_in[11];
    const float* Wh    = (const float*)d_in[12];
    const float* Uh    = (const float*)d_in[13];
    const float* bh    = (const float*)d_in[14];
    const float* W_r1  = (const float*)d_in[15];
    const float* b_r1  = (const float*)d_in[16];
    const float* W_r2  = (const float*)d_in[17];
    const float* b_r2  = (const float*)d_in[18];
    float* out = (float*)d_out;

    char* ws = (char*)d_ws;
    size_t off = 0;
    auto alloc = [&](size_t bytes) -> void* {
        void* p = (void*)(ws + off);
        off += (bytes + 255) & ~(size_t)255;
        return p;
    };

    const size_t WMAT = (size_t)256 * 256 * 2;          // bf16 256x256
    const size_t BMAT = (size_t)256 * 32 * 2;           // bf16 256x32
    const size_t ACT  = (size_t)BATCH * 256 * 256 * 2;  // per-batch activations bf16
    const size_t JPAD = (size_t)BATCH * 256 * 32 * 2;

    __bf16* WmsgT = (__bf16*)alloc(WMAT);
    __bf16* WzT   = (__bf16*)alloc(WMAT);
    __bf16* WrT   = (__bf16*)alloc(WMAT);
    __bf16* WhT   = (__bf16*)alloc(WMAT);
    __bf16* UzT   = (__bf16*)alloc(WMAT);
    __bf16* UrT   = (__bf16*)alloc(WMAT);
    __bf16* UhT   = (__bf16*)alloc(WMAT);
    __bf16* Wr1T  = (__bf16*)alloc(WMAT);
    __bf16* WzB   = (__bf16*)alloc(BMAT);
    __bf16* WrB   = (__bf16*)alloc(BMAT);
    __bf16* WhB   = (__bf16*)alloc(BMAT);
    __bf16* jetsP = (__bf16*)alloc(JPAD);
    __bf16* dadsB = (__bf16*)alloc(ACT);
    __bf16* hbuf  = (__bf16*)alloc(ACT);
    __bf16* bufA  = (__bf16*)alloc(ACT);   // tT -> rh -> g
    __bf16* bufB  = (__bf16*)alloc(ACT);   // m
    __bf16* bufC  = (__bf16*)alloc(ACT);   // z
    float*  sbuf  = (float*)alloc((size_t)BATCH * 256 * 4);

    const size_t AS = (size_t)256 * 256;  // per-batch element stride for 256x256 activations
    const size_t JS = (size_t)256 * 32;

    // ---- prep ----
    k_transposeW<<<256, 256, 0, stream>>>(W_msg, WmsgT);
    k_transposeW<<<256, 256, 0, stream>>>(Wz,    WzT);   // top 256 rows of (272,256)
    k_transposeW<<<256, 256, 0, stream>>>(Wr,    WrT);
    k_transposeW<<<256, 256, 0, stream>>>(Wh,    WhT);
    k_transposeW<<<256, 256, 0, stream>>>(Uz,    UzT);
    k_transposeW<<<256, 256, 0, stream>>>(Ur,    UrT);
    k_transposeW<<<256, 256, 0, stream>>>(Uh,    UhT);
    k_transposeW<<<256, 256, 0, stream>>>(W_r1,  Wr1T);
    k_botT<<<32, 256, 0, stream>>>(Wz, WzB);
    k_botT<<<32, 256, 0, stream>>>(Wr, WrB);
    k_botT<<<32, 256, 0, stream>>>(Wh, WhB);
    k_jets_pad<<<(BATCH * 256 * 32) / 256, 256, 0, stream>>>(jets, jetsP);
    k_embed<<<BATCH * 256, 256, 0, stream>>>(jets, W_emb, b_emb, hbuf);
    k_tobf<<<(BATCH * 256 * 256) / 256, 256, 0, stream>>>(dads, dadsB, BATCH * 256 * 256);

    const int GG = BATCH * 4;  // gemm grid

    for (int it = 0; it < NITER; ++it) {
        // t = h @ W_msg + b_msg  -> stored transposed (bufA = t^T, [H][N] per batch)
        gemm_mpnn<EPI_TSTORE, 1><<<GG, 256, 0, stream>>>(
            hbuf, AS, WmsgT, 0, 256,
            nullptr, 0, nullptr, 0, 0, nullptr, 0, nullptr, 0, 0,
            b_msg, nullptr, nullptr, 0, bufA, AS);
        // m = tanh(dads @ t)   (B operand = t^T, per-batch)
        gemm_mpnn<EPI_ROW_TANH, 1><<<GG, 256, 0, stream>>>(
            dadsB, AS, bufA, AS, 256,
            nullptr, 0, nullptr, 0, 0, nullptr, 0, nullptr, 0, 0,
            nullptr, nullptr, nullptr, 0, bufB, AS);
        // z = sigmoid(m@Wz_top + h@Uz + jets@Wz_bot + bz)
        gemm_mpnn<EPI_ROW_SIG, 3><<<GG, 256, 0, stream>>>(
            bufB, AS, WzT, 0, 256,
            hbuf, AS, UzT, 0, 256,
            jetsP, JS, WzB, 0, FPAD,
            bz, nullptr, nullptr, 0, bufC, AS);
        // rh = sigmoid(m@Wr_top + h@Ur + jets@Wr_bot + br) * h
        gemm_mpnn<EPI_ROW_SIGMULH, 3><<<GG, 256, 0, stream>>>(
            bufB, AS, WrT, 0, 256,
            hbuf, AS, UrT, 0, 256,
            jetsP, JS, WrB, 0, FPAD,
            br, hbuf, nullptr, AS, bufA, AS);
        // h = (1-z)*h + z*tanh(m@Wh_top + rh@Uh + jets@Wh_bot + bh)
        gemm_mpnn<EPI_ROW_GRU, 3><<<GG, 256, 0, stream>>>(
            bufB, AS, WhT, 0, 256,
            bufA, AS, UhT, 0, 256,
            jetsP, JS, WhB, 0, FPAD,
            bh, hbuf, bufC, AS, hbuf, AS);
    }

    // g = tanh(h @ W_r1 + b_r1)
    gemm_mpnn<EPI_ROW_TANH, 1><<<GG, 256, 0, stream>>>(
        hbuf, AS, Wr1T, 0, 256,
        nullptr, 0, nullptr, 0, 0, nullptr, 0, nullptr, 0, 0,
        b_r1, nullptr, nullptr, 0, bufA, AS);
    // s[b,k] = sum_n g[b,n,k];  out[b] = s[b] @ W_r2 + 256*b_r2
    k_reduce<<<BATCH, 256, 0, stream>>>(bufA, sbuf);
    k_final<<<BATCH, 256, 0, stream>>>(sbuf, W_r2, b_r2, out);
    (void)in_sizes; (void)n_in; (void)out_size; (void)ws_size;
}
